// MistralAttention_35759897707159
// MI455X (gfx1250) — matrix operand order repaired and checked by reading, not compiled, not run on hardware
//
#include <hip/hip_runtime.h>

typedef __attribute__((ext_vector_type(16))) __bf16 v16bf;
typedef __attribute__((ext_vector_type(8)))  float  v8f;
typedef int v4i __attribute__((vector_size(16)));
typedef unsigned int u32;

#define T_TOK 2048
#define HID 4096
#define NQH 32
#define NKVH 8
#define HD 128
#define SW 512

#if defined(__has_builtin)
#if __has_builtin(__builtin_amdgcn_global_load_async_to_lds_b128) && \
    __has_builtin(__builtin_amdgcn_s_wait_asynccnt)
#define HAVE_ASYNC_LDS 1
#endif
#endif
#ifndef HAVE_ASYNC_LDS
#define HAVE_ASYNC_LDS 0
#endif

// builtin signature (from hipcc diagnostic): param0 = v4i32 addrspace(1)*,
// param1 = v4i32 addrspace(3)*, then imm offset, imm cpol
#define AS1V4(p) ((__attribute__((address_space(1))) v4i*)(p))
#define AS3V4(p) ((__attribute__((address_space(3))) v4i*)(p))

__device__ inline __bf16 f2bf(float f) {
  union { float f; u32 u; } a; a.f = f;
  u32 u = a.u;
  unsigned short r = (unsigned short)((u + 0x7FFFu + ((u >> 16) & 1u)) >> 16);
  union { unsigned short s; __bf16 b; } o; o.s = r;
  return o.b;
}

union Frag2 { v16bf v; v4i q[2]; };
union VPack { v4i v; __bf16 h[8]; };

// ---------------------------------------------------------------------------
// elementwise f32 -> bf16
// ---------------------------------------------------------------------------
__global__ void cvt_kernel(const float* __restrict__ in, __bf16* __restrict__ out,
                           size_t n) {
  size_t i = (size_t)blockIdx.x * blockDim.x + threadIdx.x;
  if (i < n) out[i] = f2bf(in[i]);
}

// ---------------------------------------------------------------------------
// bf16 GEMM: C[M,N] f32 = A[M,K] @ B[K,N]   (M,N mult of 128, K mult of 32)
// 256 threads = 8 waves; tile 128x128; each wave 64x32 (4x2 wmma tiles).
// Double-buffered LDS: tile t+1 streamed (async b128 for A, register-staged
// b128 for B with transpose) while tile t runs on the matrix cores.
// ---------------------------------------------------------------------------
#define KT 32
#define LDT 40   // halves per LDS tile row (80B -> every b128 chunk 16B aligned)

__global__ __launch_bounds__(256) void gemm_bf16(
    const __bf16* __restrict__ A, const __bf16* __restrict__ B,
    float* __restrict__ C, int M, int N, int K) {
  __shared__ __bf16 As[2][128 * LDT];
  __shared__ __bf16 Bs[2][128 * LDT];
  const int tid = threadIdx.x;
  const int wid = tid >> 5;
  const int lane = tid & 31;
  const int l = lane & 15, hi = lane >> 4;
  const int m0 = blockIdx.y * 128;
  const int n0 = blockIdx.x * 128;
  const int waveM = wid >> 2, waveN = wid & 3;
  const int nk = K / KT;

  v4i ar[2];  // A register staging (fallback path only)
  v4i br[2];  // B register staging

  auto issueA = [&](int k0, int buf) {
#if HAVE_ASYNC_LDS
#pragma unroll
    for (int i = 0; i < 2; ++i) {
      int e = tid + i * 256;          // 0..511 b128 chunks (128 rows x 4)
      int r = e >> 2, c8 = e & 3;
      __builtin_amdgcn_global_load_async_to_lds_b128(
          AS1V4(A + (size_t)(m0 + r) * K + k0 + c8 * 8),
          AS3V4(&As[buf][r * LDT + c8 * 8]), 0, 0);
    }
#else
#pragma unroll
    for (int i = 0; i < 2; ++i) {
      int e = tid + i * 256;
      int r = e >> 2, c8 = e & 3;
      ar[i] = *(const v4i*)(A + (size_t)(m0 + r) * K + k0 + c8 * 8);
    }
#endif
    (void)buf;
  };
  auto commitA = [&](int buf) {
#if !HAVE_ASYNC_LDS
#pragma unroll
    for (int i = 0; i < 2; ++i) {
      int e = tid + i * 256;
      int r = e >> 2, c8 = e & 3;
      *(v4i*)&As[buf][r * LDT + c8 * 8] = ar[i];
    }
#endif
    (void)buf;
  };
  auto loadB = [&](int k0) {
#pragma unroll
    for (int i = 0; i < 2; ++i) {
      int e = tid + i * 256;          // 0..511 b128 chunks (32 rows x 16)
      int kk = e >> 4, c8 = e & 15;
      br[i] = *(const v4i*)(B + (size_t)(k0 + kk) * N + n0 + c8 * 8);
    }
  };
  auto commitB = [&](int buf) {       // transpose-store: Bs[n][k]
#pragma unroll
    for (int i = 0; i < 2; ++i) {
      int e = tid + i * 256;
      int kk = e >> 4, c8 = e & 15;
      VPack u; u.v = br[i];
#pragma unroll
      for (int q = 0; q < 8; ++q)
        Bs[buf][(c8 * 8 + q) * LDT + kk] = u.h[q];
    }
  };

  v8f zero = {};
  v8f acc[4][2];
#pragma unroll
  for (int i = 0; i < 4; ++i)
#pragma unroll
    for (int j = 0; j < 2; ++j) acc[i][j] = zero;

  // prologue: stage tile 0 into buffer 0
  issueA(0, 0);
  loadB(0);
  commitA(0);
  commitB(0);
#if HAVE_ASYNC_LDS
  __builtin_amdgcn_s_wait_asynccnt(0);
#endif
  __syncthreads();

  for (int t = 0; t < nk; ++t) {
    const int cur = t & 1, nxt = cur ^ 1;
    const int k1 = (t + 1) * KT;
    const bool more = (t + 1 < nk);
    if (more) {
      issueA(k1, nxt);   // async stream into the other buffer
      loadB(k1);         // b128 global loads into registers
    }
    if (t + 2 < nk) {
      __builtin_prefetch(A + (size_t)(m0 + (tid >> 1)) * K + k1 + KT + (tid & 1) * 16, 0, 1);
      __builtin_prefetch(B + (size_t)(k1 + KT + (tid >> 3)) * N + n0 + (tid & 7) * 16, 0, 1);
    }

    // compute on current buffer
    Frag2 af[4];
#pragma unroll
    for (int i = 0; i < 4; ++i) {
      const __bf16* rowp = &As[cur][(waveM * 64 + i * 16 + l) * LDT];
      af[i].q[0] = *(const v4i*)(rowp + hi * 8);
      af[i].q[1] = *(const v4i*)(rowp + 16 + hi * 8);
    }
    Frag2 bfr[2];
#pragma unroll
    for (int j = 0; j < 2; ++j) {
      const __bf16* colp = &Bs[cur][(waveN * 32 + j * 16 + l) * LDT];
      bfr[j].q[0] = *(const v4i*)(colp + hi * 8);
      bfr[j].q[1] = *(const v4i*)(colp + 16 + hi * 8);
    }
#pragma unroll
    for (int i = 0; i < 4; ++i)
#pragma unroll
      for (int j = 0; j < 2; ++j)
        acc[i][j] = __builtin_amdgcn_wmma_f32_16x16x32_bf16(
            false, af[i].v, false, bfr[j].v, (short)0, acc[i][j], false, false);

    if (more) {
      commitA(nxt);
      commitB(nxt);
    }
#if HAVE_ASYNC_LDS
    __builtin_amdgcn_s_wait_asynccnt(0);
#endif
    __syncthreads();
  }

#pragma unroll
  for (int i = 0; i < 4; ++i)
#pragma unroll
    for (int j = 0; j < 2; ++j) {
      int col = n0 + waveN * 32 + j * 16 + l;
#pragma unroll
      for (int r = 0; r < 8; ++r) {
        int row = m0 + waveM * 64 + i * 16 + hi * 8 + r;
        C[(size_t)row * N + col] = acc[i][j][r];
      }
    }
}

// ---------------------------------------------------------------------------
// RoPE: f32 Q/K -> bf16 rotated Q/K. grid = T * (NQH + NKVH), block = 64
// ---------------------------------------------------------------------------
__global__ void rope_kernel(const float* __restrict__ Qf, const float* __restrict__ Kf,
                            const int* __restrict__ cu, int nseq,
                            __bf16* __restrict__ Qb, __bf16* __restrict__ Kb) {
  int idx = blockIdx.x;
  int t = idx / (NQH + NKVH);
  int hh = idx % (NQH + NKVH);
  int i = threadIdx.x;                     // 0..63 dim pairs
  int s = 0;
  for (int j = 1; j < nseq; ++j) if (t >= cu[j]) s = j;
  int pos = t - cu[s];
  float freq = (float)pos * __powf(10000.0f, -(float)(2 * i) / 128.0f);
  float sn, c;
  __sincosf(freq, &sn, &c);
  const float* src; __bf16* dst;
  if (hh < NQH) {
    src = Qf + ((size_t)t * NQH + hh) * HD;
    dst = Qb + ((size_t)t * NQH + hh) * HD;
  } else {
    int kh = hh - NQH;
    src = Kf + ((size_t)t * NKVH + kh) * HD;
    dst = Kb + ((size_t)t * NKVH + kh) * HD;
  }
  float a = src[i], b = src[i + 64];
  dst[i]      = f2bf(a * c - b * sn);
  dst[i + 64] = f2bf(b * c + a * sn);
}

// ---------------------------------------------------------------------------
// Flash attention (GQA, causal + sliding window 512, block-diagonal by seq)
// grid = (NQH, T/64), block = 128 (4 waves, 16 query rows each)
// ---------------------------------------------------------------------------
#define LDP 72   // halves per P/V^T LDS row (144B -> b128-aligned fragments)

__global__ __launch_bounds__(128) void attn_kernel(
    const __bf16* __restrict__ Qb,   // [T, 32, 128]
    const __bf16* __restrict__ Kb,   // [T, 8, 128]
    const __bf16* __restrict__ Vb,   // [T, 8, 128]
    const int* __restrict__ cu, int nseq,
    __bf16* __restrict__ Ob,         // [T, 4096]
    int T) {
  __shared__ __bf16 Pl[4][16][LDP];  // per-wave P tile (16 rows x 64 keys)
  __shared__ __bf16 Vt[128][LDP];    // V^T tile (128 dims x 64 keys)

  const int h = blockIdx.x;
  const int qb = blockIdx.y;
  const int kvh = h >> 2;
  const int tid = threadIdx.x;
  const int wid = tid >> 5, lane = tid & 31;
  const int l = lane & 15, hi = lane >> 4;
  const int q0 = qb * 64;
  const int qrow0 = q0 + wid * 16;
  const float scale = 0.08838834764831845f;  // 1/sqrt(128)

  int qseg[8], qtok[8];
#pragma unroll
  for (int r = 0; r < 8; ++r) {
    int t = qrow0 + hi * 8 + r;
    qtok[r] = t;
    int s = 0;
    for (int j = 1; j < nseq; ++j) if (t >= cu[j]) s = j;
    qseg[r] = s;
  }

  // Q fragments: 16 rows x 128 dims as 4 A-tiles (b128 pairs)
  Frag2 qf[4];
#pragma unroll
  for (int d = 0; d < 4; ++d) {
    const __bf16* base = Qb + ((size_t)(qrow0 + l) * NQH + h) * HD + d * 32;
    qf[d].q[0] = *(const v4i*)(base + hi * 8);
    qf[d].q[1] = *(const v4i*)(base + 16 + hi * 8);
  }

  v8f zero = {};
  v8f Oacc[8];
#pragma unroll
  for (int d = 0; d < 8; ++d) Oacc[d] = zero;
  float m_run[8], l_run[8];
#pragma unroll
  for (int r = 0; r < 8; ++r) { m_run[r] = -1e30f; l_run[r] = 0.f; }

  int kb_lo = q0 - (SW - 1);
  kb_lo = kb_lo > 0 ? (kb_lo & ~63) : 0;
  int kb_hi = q0 + 64; if (kb_hi > T) kb_hi = T;

  for (int kb = kb_lo; kb < kb_hi; kb += 64) {
    __syncthreads();  // previous iteration's LDS reads done

    // stage V^T [dim][key]: b128 register staging, then b16 transpose stores
#pragma unroll
    for (int half = 0; half < 2; ++half) {
      v4i vr[4];
#pragma unroll
      for (int i = 0; i < 4; ++i) {
        int e = tid + (half * 4 + i) * 128;   // 0..1023 b128 chunks (64 x 16)
        int r = e >> 4, c8 = e & 15;
        int tok = kb + r; if (tok >= T) tok = T - 1;
        vr[i] = *(const v4i*)(Vb + ((size_t)tok * NKVH + kvh) * HD + c8 * 8);
      }
#pragma unroll
      for (int i = 0; i < 4; ++i) {
        int e = tid + (half * 4 + i) * 128;
        int r = e >> 4, c8 = e & 15;
        VPack u; u.v = vr[i];
#pragma unroll
        for (int q = 0; q < 8; ++q) Vt[c8 * 8 + q][r] = u.h[q];
      }
    }

    // S = Q K^T  (reduction over head dim, 4 wmma steps per 16-key tile)
    v8f S[4];
#pragma unroll
    for (int j = 0; j < 4; ++j) {
      v8f s = zero;
      int key = kb + j * 16 + l; if (key >= T) key = T - 1;
#pragma unroll
      for (int d = 0; d < 4; ++d) {
        Frag2 kf;
        const __bf16* base = Kb + ((size_t)key * NKVH + kvh) * HD + d * 32 + hi * 8;
        kf.q[0] = *(const v4i*)(base);
        kf.q[1] = *(const v4i*)(base + 16);
        s = __builtin_amdgcn_wmma_f32_16x16x32_bf16(
            false, qf[d].v, false, kf.v, (short)0, s, false, false);
      }
      S[j] = s;
    }

    // scale + mask
#pragma unroll
    for (int j = 0; j < 4; ++j) {
      int kt = kb + j * 16 + l;
      int ks = 0;
      for (int i = 1; i < nseq; ++i) if (kt >= cu[i]) ks = i;
#pragma unroll
      for (int r = 0; r < 8; ++r) {
        float sv = S[j][r] * scale;
        bool ok = (kt < T) && (ks == qseg[r]) && (kt <= qtok[r]) &&
                  (kt >= qtok[r] - (SW - 1));
        S[j][r] = ok ? sv : -1e30f;
      }
    }

    // online softmax per row (rows live in VGPR index; 16 lanes hold the N dim)
#pragma unroll
    for (int r = 0; r < 8; ++r) {
      float m = fmaxf(fmaxf(S[0][r], S[1][r]), fmaxf(S[2][r], S[3][r]));
      m = fmaxf(m, __shfl_xor(m, 8, 16));
      m = fmaxf(m, __shfl_xor(m, 4, 16));
      m = fmaxf(m, __shfl_xor(m, 2, 16));
      m = fmaxf(m, __shfl_xor(m, 1, 16));
      float mn = fmaxf(m_run[r], m);
      float alpha = __expf(m_run[r] - mn);
      float mu = fmaxf(mn, -1e29f);  // keeps masked exp() at 0 even if all masked
      float psum = 0.f;
#pragma unroll
      for (int j = 0; j < 4; ++j) {
        float p = __expf(S[j][r] - mu);
        psum += p;
        S[j][r] = p;
      }
      psum += __shfl_xor(psum, 8, 16);
      psum += __shfl_xor(psum, 4, 16);
      psum += __shfl_xor(psum, 2, 16);
      psum += __shfl_xor(psum, 1, 16);
      l_run[r] = l_run[r] * alpha + psum;
      m_run[r] = mn;
#pragma unroll
      for (int d = 0; d < 8; ++d) Oacc[d][r] *= alpha;
      int row = hi * 8 + r;
#pragma unroll
      for (int j = 0; j < 4; ++j)
        Pl[wid][row][j * 16 + l] = f2bf(S[j][r]);
    }

    __syncthreads();  // Pl + Vt visible

    // O += P @ V : reduction over 64 keys = 2 wmma K-steps
#pragma unroll
    for (int ks = 0; ks < 2; ++ks) {
      Frag2 pf;
      pf.q[0] = *(const v4i*)&Pl[wid][l][ks * 32 + hi * 8];
      pf.q[1] = *(const v4i*)&Pl[wid][l][ks * 32 + 16 + hi * 8];
#pragma unroll
      for (int d = 0; d < 8; ++d) {
        Frag2 vf;
        vf.q[0] = *(const v4i*)&Vt[d * 16 + l][ks * 32 + hi * 8];
        vf.q[1] = *(const v4i*)&Vt[d * 16 + l][ks * 32 + 16 + hi * 8];
        Oacc[d] = __builtin_amdgcn_wmma_f32_16x16x32_bf16(
            false, pf.v, false, vf.v, (short)0, Oacc[d], false, false);
      }
    }
  }

  // normalize + write bf16 [T, NQH*HD]
#pragma unroll
  for (int r = 0; r < 8; ++r) {
    int row = qrow0 + hi * 8 + r;
    float inv = 1.0f / l_run[r];
#pragma unroll
    for (int d = 0; d < 8; ++d)
      Ob[(size_t)row * (NQH * HD) + h * HD + d * 16 + l] = f2bf(Oacc[d][r] * inv);
  }
}

// ---------------------------------------------------------------------------
extern "C" void kernel_launch(void* const* d_in, const int* in_sizes, int n_in,
                              void* d_out, int out_size, void* d_ws, size_t ws_size,
                              hipStream_t stream) {
  const float* X  = (const float*)d_in[0];
  const float* Wq = (const float*)d_in[1];
  const float* Wk = (const float*)d_in[2];
  const float* Wv = (const float*)d_in[3];
  const float* Wo = (const float*)d_in[4];
  const int*   cu = (const int*)d_in[5];
  const int nseq = in_sizes[5] - 1;
  const int T = T_TOK;

  char* p = (char*)d_ws;
  auto alloc = [&](size_t bytes) {
    char* r = p;
    p += (bytes + 255) & ~(size_t)255;
    return r;
  };
  __bf16* Xb  = (__bf16*)alloc((size_t)T * HID * 2);
  __bf16* Wqb = (__bf16*)alloc((size_t)HID * (NQH * HD) * 2);
  __bf16* Wkb = (__bf16*)alloc((size_t)HID * (NKVH * HD) * 2);
  __bf16* Wvb = (__bf16*)alloc((size_t)HID * (NKVH * HD) * 2);
  __bf16* Wob = (__bf16*)alloc((size_t)(NQH * HD) * HID * 2);
  float*  Qf  = (float*)alloc((size_t)T * NQH * HD * 4);
  float*  Kf  = (float*)alloc((size_t)T * NKVH * HD * 4);
  float*  Vf  = (float*)alloc((size_t)T * NKVH * HD * 4);
  __bf16* Qb  = (__bf16*)alloc((size_t)T * NQH * HD * 2);
  __bf16* Kb  = (__bf16*)alloc((size_t)T * NKVH * HD * 2);
  __bf16* Vb  = (__bf16*)alloc((size_t)T * NKVH * HD * 2);
  __bf16* Ab  = (__bf16*)alloc((size_t)T * NQH * HD * 2);

  auto cvt = [&](const float* src, __bf16* dst, size_t n) {
    cvt_kernel<<<(unsigned)((n + 255) / 256), 256, 0, stream>>>(src, dst, n);
  };
  cvt(X,  Xb,  (size_t)T * HID);
  cvt(Wq, Wqb, (size_t)HID * NQH * HD);
  cvt(Wk, Wkb, (size_t)HID * NKVH * HD);
  cvt(Wv, Wvb, (size_t)HID * NKVH * HD);
  cvt(Wo, Wob, (size_t)(NQH * HD) * HID);

  // QKV projections
  gemm_bf16<<<dim3((NQH * HD) / 128, T / 128), 256, 0, stream>>>(
      Xb, Wqb, Qf, T, NQH * HD, HID);
  gemm_bf16<<<dim3((NKVH * HD) / 128, T / 128), 256, 0, stream>>>(
      Xb, Wkb, Kf, T, NKVH * HD, HID);
  gemm_bf16<<<dim3((NKVH * HD) / 128, T / 128), 256, 0, stream>>>(
      Xb, Wvb, Vf, T, NKVH * HD, HID);

  // RoPE + V convert
  rope_kernel<<<T * (NQH + NKVH), 64, 0, stream>>>(Qf, Kf, cu, nseq, Qb, Kb);
  cvt(Vf, Vb, (size_t)T * NKVH * HD);

  // flash attention
  attn_kernel<<<dim3(NQH, T / 64), 128, 0, stream>>>(Qb, Kb, Vb, cu, nseq, Ab, T);

  // output projection -> f32 d_out
  gemm_bf16<<<dim3(HID / 128, T / 128), 256, 0, stream>>>(
      Ab, Wob, (float*)d_out, T, HID, NQH * HD);
}
